// AdvancedNeuralMemory_40596030882229
// MI455X (gfx1250) — matrix operand order repaired and checked by reading, not compiled, not run on hardware
//
#include <hip/hip_runtime.h>
#include <math.h>
#include <stdint.h>

// ---------- types ----------
typedef __bf16 bf16;
typedef bf16  v8bf  __attribute__((ext_vector_type(8)));
typedef bf16  v16bf __attribute__((ext_vector_type(16)));
typedef float v8f   __attribute__((ext_vector_type(8)));

// ---------- problem constants ----------
#define B_   4
#define S_   8192
#define D_   1024
#define DM_  512
#define MTOT (B_ * S_)          // 32768 rows
#define NQKV (3 * DM_)          // 1536
#define CH_  128                // scan chunk length
#define NCH  (S_ / CH_)         // 64 chunks per batch

// ---------- TDM availability ----------
#if defined(__has_builtin)
#if __has_builtin(__builtin_amdgcn_tensor_load_to_lds) && \
    __has_builtin(__builtin_amdgcn_s_wait_tensorcnt)
#define USE_TDM 1
#else
#define USE_TDM 0
#endif
#else
#define USE_TDM 0
#endif

__device__ __forceinline__ v16bf make16(v8bf lo, v8bf hi) {
    return __builtin_shufflevector(lo, hi, 0,1,2,3,4,5,6,7,8,9,10,11,12,13,14,15);
}

#if USE_TDM
typedef unsigned int u32x4 __attribute__((ext_vector_type(4)));
typedef int          i32x4 __attribute__((ext_vector_type(4)));
typedef int          i32x8 __attribute__((ext_vector_type(8)));

// DMA one 128-row x 32-col bf16 tile (row stride = row_stride_elems) from
// global into LDS at lds_off, padding 4 DWORDs after every 16 DWORDs so the
// LDS image matches a [128][40] bf16 array. 2D descriptor per ISA 08 §8.
__device__ __forceinline__ void tdm_load_tile_128x32_bf16(
    const bf16* gsrc, unsigned lds_off, int row_stride_elems, int nrows_total)
{
    unsigned long long ga = (unsigned long long)(uintptr_t)gsrc;
    u32x4 g0;
    g0[0] = 1u;                                   // count=1, user descriptor
    g0[1] = lds_off;                              // D#.lds_addr
    g0[2] = (unsigned)(ga & 0xffffffffu);         // global_addr[31:0]
    g0[3] = (unsigned)((ga >> 32) & 0x01ffffffu)  // global_addr[56:32]
          | (2u << 30);                           // type = 2 ("image")
    i32x8 g1;
    g1[0] = (int)((1u << 16)     // data_size = 2 bytes
                | (1u << 20)     // pad_enable
                | (3u << 22)     // pad_interval: 16 DWORDs
                | (3u << 25));   // pad_amount:   4 DWORDs
    g1[1] = (int)(((unsigned)row_stride_elems & 0xffffu) << 16);     // tensor_dim0[15:0]
    g1[2] = (int)(((unsigned)row_stride_elems >> 16)                 // tensor_dim0[31:16]
                | (((unsigned)nrows_total & 0xffffu) << 16));        // tensor_dim1[15:0]
    g1[3] = (int)(((unsigned)nrows_total >> 16) | (32u << 16));      // dim1 hi | tile_dim0=32
    g1[4] = 128;                                  // tile_dim1 = 128 rows (tile_dim2 = 0)
    g1[5] = (int)(unsigned)row_stride_elems;      // tensor_dim0_stride[31:0]
    g1[6] = 0;                                    // stride hi | tensor_dim1_stride lo
    g1[7] = 0;
    i32x4 z4 = {0, 0, 0, 0};
#if __clang_major__ >= 23
    i32x8 z8 = {0, 0, 0, 0, 0, 0, 0, 0};
    __builtin_amdgcn_tensor_load_to_lds(g0, g1, z4, z4, z8, 0);
#else
    __builtin_amdgcn_tensor_load_to_lds(g0, g1, z4, z4, 0);
#endif
}
#endif // USE_TDM

// =====================================================================
// Generic WMMA GEMM:  C[M,N] = A[M,K] * W[N,K]^T + bias[N]
// Block = 256 threads = 8 waves; tile 128(M) x 128(N); K-step 32 (bf16).
// Wave w owns rows [16w,16w+16) x all 128 cols (8 x v8f accumulators).
// TDM path: double-buffered LDS tiles; wave 0 issues tile i+1 then waits
// tensorcnt<= (#ops just issued), so tile i is ready while i+1 streams in
// behind the WMMA compute. fp32-A staged via VALU bf16 conversion.
// =====================================================================
template <bool AF32, bool OBF16>
__global__ __launch_bounds__(256) void gemm_wmma_kernel(
    const void* __restrict__ Ap, const bf16* __restrict__ W,
    const float* __restrict__ bias, void* __restrict__ Cp,
    int M, int N, int K)
{
    __shared__ bf16 As[2][128][40];   // 32 K + 8 pad (bank spread / TDM pad)
    __shared__ bf16 Bs[2][128][40];

    const int tid  = threadIdx.x;
    const int lane = tid & 31;
    const int wv   = tid >> 5;
    const int half = lane >> 4;    // 0: lanes 0-15, 1: lanes 16-31
    const int r16  = lane & 15;
    const int tileN = blockIdx.x * 128;
    const int tileM = blockIdx.y * 128;

    const v8f vzero = {0.f,0.f,0.f,0.f,0.f,0.f,0.f,0.f};
    v8f acc[8];
#pragma unroll
    for (int i = 0; i < 8; ++i) acc[i] = vzero;

    // staging map: each thread fills 16 contiguous K of one row
    const int srow = tid >> 1;
    const int scb  = (tid & 1) * 16;

    const int nk = K / 32;

#if USE_TDM
    // ---- prologue: DMA tile 0 into buffer 0 ----
    if (wv == 0) {
        if (!AF32)
            tdm_load_tile_128x32_bf16((const bf16*)Ap + (size_t)tileM * K,
                                      (unsigned)(uintptr_t)&As[0][0][0], K, M);
        tdm_load_tile_128x32_bf16(W + (size_t)tileN * K,
                                  (unsigned)(uintptr_t)&Bs[0][0][0], K, N);
    }
#endif

    for (int it = 0; it < nk; ++it) {
        const int k0  = it * 32;
        const int cur = it & 1;
        const int nxt = cur ^ 1;

        // all waves done reading the buffers about to be (re)written
        __syncthreads();

        // ---- stage A tile (fp32 path: same-iteration VALU conversion) ----
        if (AF32) {
            const float* A = (const float*)Ap;
            const float4* src =
                (const float4*)(A + (size_t)(tileM + srow) * K + k0 + scb);
            bf16 tmp[16];
#pragma unroll
            for (int j = 0; j < 4; ++j) {
                float4 f = src[j];
                tmp[4*j+0] = (bf16)f.x; tmp[4*j+1] = (bf16)f.y;
                tmp[4*j+2] = (bf16)f.z; tmp[4*j+3] = (bf16)f.w;
            }
            *(v8bf*)&As[0][srow][scb]     = *(v8bf*)&tmp[0];
            *(v8bf*)&As[0][srow][scb + 8] = *(v8bf*)&tmp[8];
            if (k0 + 32 < K)
                __builtin_prefetch(A + (size_t)(tileM + srow) * K + k0 + 32 + scb, 0, 0);
        }

#if USE_TDM
        if (wv == 0) {
            if (it + 1 < nk) {
                // issue next tile into the other buffer, then wait until only
                // those just-issued ops are outstanding -> current tile ready
                if (!AF32)
                    tdm_load_tile_128x32_bf16(
                        (const bf16*)Ap + (size_t)tileM * K + k0 + 32,
                        (unsigned)(uintptr_t)&As[nxt][0][0], K, M);
                tdm_load_tile_128x32_bf16(
                    W + (size_t)tileN * K + k0 + 32,
                    (unsigned)(uintptr_t)&Bs[nxt][0][0], K, N);
                __builtin_amdgcn_s_wait_tensorcnt(AF32 ? 1 : 2);
            } else {
                __builtin_amdgcn_s_wait_tensorcnt(0);
            }
        }
#else
        if (!AF32) {
            const bf16* A = (const bf16*)Ap;
            const v8bf* src =
                (const v8bf*)(A + (size_t)(tileM + srow) * K + k0 + scb);
            *(v8bf*)&As[0][srow][scb]     = src[0];
            *(v8bf*)&As[0][srow][scb + 8] = src[1];
            if (k0 + 32 < K)
                __builtin_prefetch(A + (size_t)(tileM + srow) * K + k0 + 32 + scb, 0, 0);
        }
        {
            const v8bf* src =
                (const v8bf*)(W + (size_t)(tileN + srow) * K + k0 + scb);
            *(v8bf*)&Bs[0][srow][scb]     = src[0];
            *(v8bf*)&Bs[0][srow][scb + 8] = src[1];
            if (k0 + 32 < K)
                __builtin_prefetch(W + (size_t)(tileN + srow) * K + k0 + 32 + scb, 0, 0);
        }
#endif
        __syncthreads();

        // ---- pick buffers for this iteration ----
#if USE_TDM
        const bf16 (*Ause)[40] = AF32 ? As[0] : As[cur];
        const bf16 (*Buse)[40] = Bs[cur];
#else
        const bf16 (*Ause)[40] = As[0];
        const bf16 (*Buse)[40] = Bs[0];
        (void)cur; (void)nxt;
#endif

        // ---- A fragment: lane holds row r16, K halves per ISA layout ----
        const int arow = wv * 16 + r16;
        v8bf alo = *(const v8bf*)&Ause[arow][half * 8];
        v8bf ahi = *(const v8bf*)&Ause[arow][16 + half * 8];
        v16bf afrag = make16(alo, ahi);

        // ---- batch all 8 B fragments, then run WMMAs back-to-back ----
        v16bf bfr[8];
#pragma unroll
        for (int sub = 0; sub < 8; ++sub) {
            const int brow = sub * 16 + r16;
            bfr[sub] = make16(*(const v8bf*)&Buse[brow][half * 8],
                              *(const v8bf*)&Buse[brow][16 + half * 8]);
        }
#pragma unroll
        for (int sub = 0; sub < 8; ++sub)
            acc[sub] = __builtin_amdgcn_wmma_f32_16x16x32_bf16(
                false, afrag, false, bfr[sub], (short)0, acc[sub], false, false);
    }

    // ---- epilogue: C layout lane L col=r16, VGPR r -> row r (+8 if half) ----
#pragma unroll
    for (int sub = 0; sub < 8; ++sub) {
        const int col  = tileN + sub * 16 + r16;
        const float bc = bias[col];
#pragma unroll
        for (int r = 0; r < 8; ++r) {
            const int row = tileM + wv * 16 + (half ? r + 8 : r);
            float v = acc[sub][r] + bc;
            if (OBF16)
                ((bf16*)Cp)[(size_t)row * N + col] = (bf16)v;
            else
                ((float*)Cp)[(size_t)row * N + col] = v;
        }
    }
}

// =====================================================================
// fp32 -> bf16 conversion (weights)
// =====================================================================
__global__ void cvt_f32_bf16_kernel(const float* __restrict__ src,
                                    bf16* __restrict__ dst, int n)
{
    int i = blockIdx.x * blockDim.x + threadIdx.x;
    if (i < n) dst[i] = (bf16)src[i];
}

// =====================================================================
// LayerNorm q (in-place bf16) & k (not stored) + u = lr * k_ln * v  (fp32)
// One wave32 per row of qkv [MTOT, 1536]; Dm=512 -> 16 elems/lane.
// =====================================================================
__global__ __launch_bounds__(256) void ln_qkv_kernel(
    bf16* __restrict__ qkv,
    const float* __restrict__ qn_g, const float* __restrict__ qn_b,
    const float* __restrict__ kn_g, const float* __restrict__ kn_b,
    const float* __restrict__ lr_p, float* __restrict__ u)
{
    const int tid  = threadIdx.x;
    const int lane = tid & 31;
    const int row  = blockIdx.x * 8 + (tid >> 5);
    bf16* base = qkv + (size_t)row * NQKV;
    const float lr = lr_p[0];

    // ---- q ----
    float qv[16], s = 0.f, sq = 0.f;
#pragma unroll
    for (int i = 0; i < 16; ++i) {
        float f = (float)base[lane + 32 * i];
        qv[i] = f; s += f; sq += f * f;
    }
#pragma unroll
    for (int off = 16; off >= 1; off >>= 1) {
        s  += __shfl_xor(s,  off);
        sq += __shfl_xor(sq, off);
    }
    float mu = s * (1.f / DM_);
    float rs = rsqrtf(sq * (1.f / DM_) - mu * mu + 1e-5f);
#pragma unroll
    for (int i = 0; i < 16; ++i) {
        int c = lane + 32 * i;
        base[c] = (bf16)((qv[i] - mu) * rs * qn_g[c] + qn_b[c]);
    }

    // ---- k ----
    float kv[16]; s = 0.f; sq = 0.f;
#pragma unroll
    for (int i = 0; i < 16; ++i) {
        float f = (float)base[DM_ + lane + 32 * i];
        kv[i] = f; s += f; sq += f * f;
    }
#pragma unroll
    for (int off = 16; off >= 1; off >>= 1) {
        s  += __shfl_xor(s,  off);
        sq += __shfl_xor(sq, off);
    }
    mu = s * (1.f / DM_);
    rs = rsqrtf(sq * (1.f / DM_) - mu * mu + 1e-5f);

    // ---- u = lr * k_ln * v ----
#pragma unroll
    for (int i = 0; i < 16; ++i) {
        int c = lane + 32 * i;
        float kn = (kv[i] - mu) * rs * kn_g[c] + kn_b[c];
        float vv = (float)base[2 * DM_ + c];
        u[(size_t)row * DM_ + c] = lr * kn * vv;
    }
}

// =====================================================================
// Gated scan, constant gate g = 1 - ff.
// 4a: per-chunk local scan in place, save chunk-final as raw carry.
// 4b: sequential carry combine -> exclusive carry-in (in place).
// 4c: mem = local + g^(sl+1)*carry_in; p = q * mem (bf16).
// =====================================================================
__global__ void scan_local_kernel(float* __restrict__ u,
                                  float* __restrict__ carry,
                                  const float* __restrict__ ff_p)
{
    const int m  = blockIdx.x * blockDim.x + threadIdx.x;   // 0..511
    const int ch = blockIdx.y;                              // 0..NCH-1
    const int b  = blockIdx.z;                              // 0..3
    const float g = 1.f - ff_p[0];
    size_t base = ((size_t)(b * S_ + ch * CH_)) * DM_ + m;
    float acc = 0.f;
    for (int i = 0; i < CH_; ++i) {
        float x = u[base + (size_t)i * DM_];
        acc = g * acc + x;
        u[base + (size_t)i * DM_] = acc;
    }
    carry[((size_t)(b * NCH + ch)) * DM_ + m] = acc;
}

__global__ void carry_fix_kernel(float* __restrict__ carry,
                                 const float* __restrict__ ff_p)
{
    const int id = blockIdx.x * blockDim.x + threadIdx.x;   // 0..2047
    const int b = id >> 9, m = id & 511;
    const float g  = 1.f - ff_p[0];
    const float gL = powf(g, (float)CH_);
    float c = 0.f;
    for (int ch = 0; ch < NCH; ++ch) {
        size_t idx = ((size_t)(b * NCH + ch)) * DM_ + m;
        float raw = carry[idx];
        carry[idx] = c;            // exclusive carry-in for chunk ch
        c = gL * c + raw;
    }
}

__global__ void apply_kernel(const float* __restrict__ u,
                             const float* __restrict__ carry,
                             const bf16* __restrict__ qkv,
                             bf16* __restrict__ p,
                             const float* __restrict__ ff_p)
{
    const size_t idx = (size_t)blockIdx.x * blockDim.x + threadIdx.x;
    const int m   = (int)(idx & (DM_ - 1));
    const int row = (int)(idx >> 9);
    const int b = row >> 13, sidx = row & (S_ - 1);
    const int ch = sidx >> 7, sl = sidx & (CH_ - 1);
    const float g = 1.f - ff_p[0];
    float mem = u[idx] +
        powf(g, (float)(sl + 1)) * carry[((size_t)(b * NCH + ch)) * DM_ + m];
    float q = (float)qkv[(size_t)row * NQKV + m];
    p[idx] = (bf16)(q * mem);
}

// =====================================================================
// Host-side launch
// =====================================================================
extern "C" void kernel_launch(void* const* d_in, const int* in_sizes, int n_in,
                              void* d_out, int out_size, void* d_ws, size_t ws_size,
                              hipStream_t stream)
{
    const float* x    = (const float*)d_in[0];
    const float* Wd   = (const float*)d_in[1];
    const float* bd   = (const float*)d_in[2];
    const float* Wq   = (const float*)d_in[3];
    const float* bq   = (const float*)d_in[4];
    const float* Wk   = (const float*)d_in[5];
    const float* bk   = (const float*)d_in[6];
    const float* Wv   = (const float*)d_in[7];
    const float* bv   = (const float*)d_in[8];
    const float* Wu   = (const float*)d_in[9];
    const float* bu   = (const float*)d_in[10];
    const float* qn_g = (const float*)d_in[11];
    const float* qn_b = (const float*)d_in[12];
    const float* kn_g = (const float*)d_in[13];
    const float* kn_b = (const float*)d_in[14];
    const float* lr   = (const float*)d_in[15];
    const float* ff   = (const float*)d_in[16];

    char* ws = (char*)d_ws;
    // workspace layout (bytes)
    const size_t OFF_WD   = 0;                                 // 512x1024 bf16 = 1 MB
    const size_t OFF_WQKV = OFF_WD   + (size_t)DM_ * D_ * 2;   // 1536x512 bf16 = 1.5 MB
    const size_t OFF_WU   = OFF_WQKV + (size_t)NQKV * DM_ * 2; // 1024x512 bf16 = 1 MB
    const size_t OFF_BQKV = OFF_WU   + (size_t)D_ * DM_ * 2;   // 1536 f32
    const size_t OFF_H    = OFF_BQKV + 8192;                   // h / p: 32768x512 bf16 = 32 MB
    const size_t OFF_QKV  = OFF_H    + (size_t)MTOT * DM_ * 2; // 32768x1536 bf16 = 96 MB
    const size_t OFF_U    = OFF_QKV  + (size_t)MTOT * NQKV * 2;// 32768x512 f32 = 64 MB
    const size_t OFF_CARRY= OFF_U    + (size_t)MTOT * DM_ * 4; // 4*64*512 f32 = 512 KB

    bf16*  wWd   = (bf16*)(ws + OFF_WD);
    bf16*  wQKV  = (bf16*)(ws + OFF_WQKV);
    bf16*  wWu   = (bf16*)(ws + OFF_WU);
    float* bqkv  = (float*)(ws + OFF_BQKV);
    bf16*  hbuf  = (bf16*)(ws + OFF_H);      // reused as p after QKV GEMM
    bf16*  qkv   = (bf16*)(ws + OFF_QKV);
    float* ubuf  = (float*)(ws + OFF_U);
    float* carry = (float*)(ws + OFF_CARRY);

    // ---- weight conversion fp32 -> bf16 (concat Wq|Wk|Wv along N) ----
    cvt_f32_bf16_kernel<<<(DM_*D_)/256, 256, 0, stream>>>(Wd, wWd, DM_*D_);
    cvt_f32_bf16_kernel<<<(DM_*DM_)/256, 256, 0, stream>>>(Wq, wQKV, DM_*DM_);
    cvt_f32_bf16_kernel<<<(DM_*DM_)/256, 256, 0, stream>>>(Wk, wQKV + (size_t)DM_*DM_, DM_*DM_);
    cvt_f32_bf16_kernel<<<(DM_*DM_)/256, 256, 0, stream>>>(Wv, wQKV + (size_t)2*DM_*DM_, DM_*DM_);
    cvt_f32_bf16_kernel<<<(D_*DM_)/256, 256, 0, stream>>>(Wu, wWu, D_*DM_);
    hipMemcpyAsync(bqkv,            bq, DM_*sizeof(float), hipMemcpyDeviceToDevice, stream);
    hipMemcpyAsync(bqkv + DM_,      bk, DM_*sizeof(float), hipMemcpyDeviceToDevice, stream);
    hipMemcpyAsync(bqkv + 2*DM_,    bv, DM_*sizeof(float), hipMemcpyDeviceToDevice, stream);

    // ---- 1) h = x * Wd^T + bd   [32768,512] bf16 ----
    gemm_wmma_kernel<true, true><<<dim3(DM_/128, MTOT/128), 256, 0, stream>>>(
        x, wWd, bd, hbuf, MTOT, DM_, D_);

    // ---- 2) qkv = h * Wqkv^T + bqkv   [32768,1536] bf16 ----
    gemm_wmma_kernel<false, true><<<dim3(NQKV/128, MTOT/128), 256, 0, stream>>>(
        hbuf, wQKV, bqkv, qkv, MTOT, NQKV, DM_);

    // ---- 3) LN(q) (in place), LN(k) + u = lr*k*v ----
    ln_qkv_kernel<<<MTOT/8, 256, 0, stream>>>(qkv, qn_g, qn_b, kn_g, kn_b, lr, ubuf);

    // ---- 4) gated scan ----
    scan_local_kernel<<<dim3(DM_/256, NCH, B_), 256, 0, stream>>>(ubuf, carry, ff);
    carry_fix_kernel<<<(B_*DM_)/256, 256, 0, stream>>>(carry, ff);
    apply_kernel<<<(size_t)MTOT*DM_/256, 256, 0, stream>>>(ubuf, carry, qkv, hbuf, ff);

    // ---- 5) out = p * Wu^T + bu   [32768,1024] fp32 ----
    gemm_wmma_kernel<false, false><<<dim3(D_/128, MTOT/128), 256, 0, stream>>>(
        hbuf, wWu, bu, d_out, MTOT, D_, DM_);

    (void)in_sizes; (void)n_in; (void)out_size; (void)ws_size;
}
